// STIDGCN_dynamic_patch_graph_31748398252685
// MI455X (gfx1250) — compile-verified, matrix operand order checked by reading, not run on hardware
//
#include <hip/hip_runtime.h>

// ---------------------------------------------------------------------------
// STIDGCN for MI455X (gfx1250).  Contractions via v_wmma_f32_16x16x32_bf16.
// Round 5: TDM-staged A tiles in BOTH matmul kernels (tensor_load_to_lds +
// s_wait_tensorcnt), double-buffered single-barrier pipelines, scalar wave
// gating via readfirstlane, and magic-multiply division (no v_rcp ladders).
// ---------------------------------------------------------------------------

typedef unsigned short u16;
typedef __attribute__((ext_vector_type(16))) __bf16 v16bf;
typedef __attribute__((ext_vector_type(8)))  float  v8f;
typedef __attribute__((ext_vector_type(4)))  unsigned int u32x4;
typedef __attribute__((ext_vector_type(8)))  int i32x8;
typedef __attribute__((ext_vector_type(4)))  int i32x4;

union Frag { v16bf v; unsigned int u[8]; };

__device__ __forceinline__ u16 f2bf(float f) {
    unsigned int u = __float_as_uint(f);
    u += 0x7fffu + ((u >> 16) & 1u);            // round-to-nearest-even
    return (u16)(u >> 16);
}

// exact for 0 <= n < 2^20 and divisors used here (mul = ceil(2^40/d))
__device__ __forceinline__ int qdiv40(int n, long long mul) {
    return (int)(((unsigned long long)(unsigned)n * (unsigned long long)mul) >> 40);
}

#define NEG_SLOPE 0.01f
#define TM 32
#define TN 64
#define LP 40    // LDS row pitch (u16): 64B data + 16B pad = 80B rows
#define CC 128   // channels
#define NP 192   // padded node count (170 -> 192): all tiles full
#define NREAL 170

// One 32x32 bf16 tile, global (row stride ldA elements) -> LDS rows of LP u16.
// TDM descriptor per CDNA5 ISA ch.8: data_size=2B, pad 4 dwords per 16 dwords.
__device__ __forceinline__ void tdm_a_tile(const u16* g, const void* lds, int ldA) {
    unsigned long long ga = (unsigned long long)(const void*)g;
    unsigned l = (unsigned)(unsigned long long)lds;
    u32x4 g0 = { 1u, l, (unsigned)ga,
                 ((unsigned)((ga >> 32) & 0x01FFFFFFull)) | 0x80000000u };
    i32x8 g1 = { (int)((1u << 16) | (1u << 20) | (3u << 22) | (3u << 25)),
                 (int)(32u << 16), (int)(32u << 16), (int)(32u << 16),
                 32, ldA, 0, 0 };
    i32x4 z4 = { 0, 0, 0, 0 };
#if defined(__clang_major__) && (__clang_major__ >= 23)
    i32x8 z8 = { 0, 0, 0, 0, 0, 0, 0, 0 };
    __builtin_amdgcn_tensor_load_to_lds(g0, g1, z4, z4, z8, 0);
#else
    __builtin_amdgcn_tensor_load_to_lds(g0, g1, z4, z4, 0);
#endif
}

// ---------------------------------------------------------------------------
// Strided-batch bf16 WMMA GEMM, full tiles only.  A row-major [.][ldA] (TDM-
// staged).  B column col = q*L2 + r, offset q*sBq + k*sBk + r (reg-staged).
// ---------------------------------------------------------------------------
__global__ __launch_bounds__(256) void k_wmma_gemm(
    const u16* __restrict__ A, long sAb, int ldA,
    const u16* __restrict__ Bm, long sBb, int sBq, int sBk, int L2, long l2mul,
    float* __restrict__ Cf, u16* __restrict__ Cb,
    long sCb, int sCm, int sCq,
    const float* __restrict__ bias,
    int K, int act, int accum)
{
    __shared__ __align__(16) u16 sA[2][TM][LP];
    __shared__ __align__(16) u16 sB[2][TN][LP];

    const int b    = blockIdx.z;
    const int row0 = blockIdx.y * TM;
    const int col0 = blockIdx.x * TN;
    const int tid  = threadIdx.x;
    const int lane = tid & 31;
    const int wave = tid >> 5;
    const int sw   = __builtin_amdgcn_readfirstlane(tid) >> 5;   // scalar wave id
    const int wr   = wave >> 2;
    const int wc   = wave & 3;
    const int n16  = lane & 15;
    const int kh   = lane >> 4;

    const u16* Abase = A + (long)b * sAb + (long)row0 * ldA;   // block-uniform
    const u16* Bb    = Bm + (long)b * sBb;

    // B staging geometry: 8 columns x 1 k per thread
    const int bn = (tid & 7) << 3;
    const int bk = tid >> 3;
    const bool bvec = ((L2 & 7) == 0) && ((sBk & 7) == 0) && ((sBq & 7) == 0); // uniform
    int boff[8];
    if (bvec) {
        int c0 = col0 + bn;
        int q = qdiv40(c0, l2mul), r = c0 - q * L2;
        boff[0] = q * sBq + r + bk * sBk;
    } else {
        #pragma unroll
        for (int j = 0; j < 8; ++j) {
            int col = col0 + bn + j;
            int q = qdiv40(col, l2mul), r = col - q * L2;
            boff[j] = q * sBq + r + bk * sBk;
        }
    }

    auto stageA = [&](int k0, int buf) {
        if (sw == 0) tdm_a_tile(Abase + k0, &sA[buf][0][0], ldA);
    };
    auto stageB = [&](int k0, int buf) {
        if (bvec) {
            uint4 v = *(const uint4*)(Bb + boff[0] + k0 * sBk);
            unsigned int w0 = v.x, w1 = v.y, w2 = v.z, w3 = v.w;
            sB[buf][bn + 0][bk] = (u16)w0;  sB[buf][bn + 1][bk] = (u16)(w0 >> 16);
            sB[buf][bn + 2][bk] = (u16)w1;  sB[buf][bn + 3][bk] = (u16)(w1 >> 16);
            sB[buf][bn + 4][bk] = (u16)w2;  sB[buf][bn + 5][bk] = (u16)(w2 >> 16);
            sB[buf][bn + 6][bk] = (u16)w3;  sB[buf][bn + 7][bk] = (u16)(w3 >> 16);
        } else {
            #pragma unroll
            for (int j = 0; j < 8; ++j)
                sB[buf][bn + j][bk] = Bb[boff[j] + k0 * sBk];
        }
    };

    // prologue
    stageA(0, 0);
    stageB(0, 0);
    __builtin_amdgcn_s_wait_tensorcnt(0);     // free on non-issuing waves
    __syncthreads();

    v8f acc = {};
    int buf = 0;

    for (int k0 = 0; k0 < K; k0 += 32) {
        Frag fa, fb;
        #pragma unroll
        for (int p = 0; p < 8; ++p) {
            int k = ((p & 3) << 1) + ((p >= 4) ? 16 : 0) + (kh << 3);
            fa.u[p] = *(const unsigned int*)&sA[buf][wr * 16 + n16][k];
            fb.u[p] = *(const unsigned int*)&sB[buf][wc * 16 + n16][k];
        }
        const bool nx = (k0 + 32) < K;
        if (nx) { stageA(k0 + 32, buf ^ 1); stageB(k0 + 32, buf ^ 1); }

        acc = __builtin_amdgcn_wmma_f32_16x16x32_bf16(
            false, fa.v, false, fb.v, (short)0, acc, false, false);

        __builtin_amdgcn_s_wait_tensorcnt(0);
        __syncthreads();
        buf ^= 1;
    }

    // epilogue: no guards (full tiles)
    int col = col0 + wc * 16 + n16;
    int q = qdiv40(col, l2mul), r = col - q * L2;
    int obase = q * sCq + r;
    const long cb = (long)b * sCb;
    #pragma unroll
    for (int t = 0; t < 8; ++t) {
        int m = row0 + wr * 16 + t + kh * 8;
        long o = cb + obase + m * sCm;
        float v = acc[t];
        if (accum) v += Cf[o];
        if (bias) v += bias[m];
        if (act == 1) v = (v >= 0.f) ? v : NEG_SLOPE * v;
        else if (act == 2) v = tanhf(v);
        if (Cf) Cf[o] = v;
        if (Cb) Cb[o] = f2bf(v);
    }
}

// ---------------------------------------------------------------------------
// Temporal conv (VALID, optional edge clamp) as implicit WMMA GEMM.
// Weights pre-packed [t][CC][CC] (TDM-staged); X padded [b][CC][NP][Lin].
// Double-buffered, flattened (t,k0) pipeline, one barrier per step.
// ---------------------------------------------------------------------------
__global__ __launch_bounds__(256) void k_wmma_conv(
    const u16* __restrict__ Wt,    // [KT][CC][CC]
    const u16* __restrict__ X,     // [b][CC][NP][Lin]
    float* __restrict__ Cf, u16* __restrict__ Cb,
    const float* __restrict__ bias,
    int Lin, int Lout, long loutmul, int KT, int toff, int clampF, int act)
{
    __shared__ __align__(16) u16 sA[2][TM][LP];
    __shared__ __align__(16) u16 sB[2][TN][LP];

    const int b    = blockIdx.z;
    const int row0 = blockIdx.y * TM;
    const int col0 = blockIdx.x * TN;
    const int tid  = threadIdx.x;
    const int lane = tid & 31;
    const int wave = tid >> 5;
    const int sw   = __builtin_amdgcn_readfirstlane(tid) >> 5;
    const int wr   = wave >> 2;
    const int wc   = wave & 3;
    const int n16  = lane & 15;
    const int kh   = lane >> 4;
    const int Ncol  = NP * Lout;
    const int NpLin = NP * Lin;

    const u16* Xb = X + (long)b * CC * NpLin;
    const u16* Wb = Wt + (long)row0 * CC;

    const int bn = (tid & 7) << 3;
    const int bk = tid >> 3;
    int nb[8], lj[8];
    #pragma unroll
    for (int j = 0; j < 8; ++j) {
        int col = col0 + bn + j;
        int n = qdiv40(col, loutmul);
        nb[j] = n * Lin + bk * NpLin;
        lj[j] = col - n * Lout;
    }

    const int iters = KT * (CC / 32);

    auto stageA = [&](int it, int buf) {
        if (sw == 0) {
            int t = it >> 2, k0 = (it & 3) << 5;
            tdm_a_tile(Wb + (long)t * CC * CC + k0, &sA[buf][0][0], CC);
        }
    };
    auto stageB = [&](int it, int buf) {
        int t = it >> 2, k0 = (it & 3) << 5;
        int kb = k0 * NpLin;
        #pragma unroll
        for (int j = 0; j < 8; ++j) {
            int ls = lj[j] + t + toff;
            if (clampF) ls = ls < 0 ? 0 : (ls >= Lin ? Lin - 1 : ls);
            sB[buf][bn + j][bk] = Xb[kb + nb[j] + ls];
        }
    };

    stageA(0, 0);
    stageB(0, 0);
    __builtin_amdgcn_s_wait_tensorcnt(0);
    __syncthreads();

    v8f acc = {};
    int buf = 0;

    for (int it = 0; it < iters; ++it) {
        Frag fa, fb;
        #pragma unroll
        for (int p = 0; p < 8; ++p) {
            int k = ((p & 3) << 1) + ((p >= 4) ? 16 : 0) + (kh << 3);
            fa.u[p] = *(const unsigned int*)&sA[buf][wr * 16 + n16][k];
            fb.u[p] = *(const unsigned int*)&sB[buf][wc * 16 + n16][k];
        }
        const bool nx = (it + 1) < iters;
        if (nx) { stageA(it + 1, buf ^ 1); stageB(it + 1, buf ^ 1); }

        acc = __builtin_amdgcn_wmma_f32_16x16x32_bf16(
            false, fa.v, false, fb.v, (short)0, acc, false, false);

        __builtin_amdgcn_s_wait_tensorcnt(0);
        __syncthreads();
        buf ^= 1;
    }

    int col = col0 + wc * 16 + n16;
    const long cb = (long)b * CC * Ncol;
    #pragma unroll
    for (int t = 0; t < 8; ++t) {
        int m = row0 + wr * 16 + t + kh * 8;
        long o = cb + (long)m * Ncol + col;
        float v = acc[t];
        if (bias) v += bias[m];
        if (act == 1) v = (v >= 0.f) ? v : NEG_SLOPE * v;
        else if (act == 2) v = tanhf(v);
        if (Cf) Cf[o] = v;
        if (Cb) Cb[o] = f2bf(v);
    }
}

// ---------------------------------------------------------------------------
// Conversion / repacking kernels
// ---------------------------------------------------------------------------
__global__ void k_f32_to_bf16(const float* __restrict__ in, u16* __restrict__ out, long n) {
    long i = (long)blockIdx.x * blockDim.x + threadIdx.x;
    long st = (long)gridDim.x * blockDim.x;
    for (; i < n; i += st) out[i] = f2bf(in[i]);
}

// conv weights: blocks x [m][k][t] -> blocks x [t][m][k]
__global__ void k_conv_w(const float* __restrict__ in, u16* __restrict__ out,
                         int blocks, int M, int K, int KT) {
    long n = (long)blocks * M * K * KT;
    long i = (long)blockIdx.x * blockDim.x + threadIdx.x;
    long st = (long)gridDim.x * blockDim.x;
    for (; i < n; i += st) {
        int  t  = (int)(i % KT);
        long r1 = i / KT;
        int  k  = (int)(r1 % K);
        long r2 = r1 / K;
        int  m  = (int)(r2 % M);
        long bl = r2 / M;
        out[(bl * KT + t) * (long)M * K + (long)m * K + k] = f2bf(in[i]);
    }
}

// adjacency: batches x [k][m] (NREAL x NREAL) -> batches x [m][k] (NP x NP, zero pad)
__global__ void k_adj_t(const float* __restrict__ in, u16* __restrict__ out, int batches) {
    long n = (long)batches * NP * NP;
    long i = (long)blockIdx.x * blockDim.x + threadIdx.x;
    long st = (long)gridDim.x * blockDim.x;
    for (; i < n; i += st) {
        int  k  = (int)(i % NP);
        long r1 = i / NP;
        int  m  = (int)(r1 % NP);
        long bl = r1 / NP;
        bool ok = (m < NREAL) && (k < NREAL);
        long io = ok ? (bl * NREAL * NREAL + (long)k * NREAL + m) : 0;
        float v = in[io];
        out[i] = ok ? f2bf(v) : (u16)0;
    }
}

// emb [C][NREAL][Lh] -> padded [C][NP][Lh] (zero pad)
__global__ void k_pad_emb(const float* __restrict__ in, float* __restrict__ out, int Lh) {
    const int plane = NP * Lh;
    const int c = blockIdx.y;
    for (int i = blockIdx.x * 256 + threadIdx.x; i < plane; i += gridDim.x * 256) {
        int n = i / Lh, l = i - n * Lh;
        bool ok = (n < NREAL);
        int io = ok ? ((c * NREAL + n) * Lh + l) : 0;
        float v = in[io];
        out[(long)c * plane + i] = ok ? v : 0.f;
    }
}

// ---------------------------------------------------------------------------
// Elementwise glue (per-plane grids; grid.y = B*C)
// ---------------------------------------------------------------------------
__global__ void k_split(const float* __restrict__ in, int inN, int Lx,
                        float* __restrict__ ef, u16* __restrict__ eb,
                        float* __restrict__ of, u16* __restrict__ ob) {
    const int Lh = Lx >> 1;
    const int plane = NP * Lh;
    const float* inp = in + (long)blockIdx.y * inN * Lx;
    const long o0 = (long)blockIdx.y * plane;
    for (int i = blockIdx.x * 256 + threadIdx.x; i < plane; i += gridDim.x * 256) {
        int n = i / Lh, l = i - n * Lh;
        bool ok = (n < inN);
        int io = ok ? (n * Lx + 2 * l) : 0;
        float ve = inp[io], vo = inp[io + 1];
        ve = ok ? ve : 0.f;  vo = ok ? vo : 0.f;
        ef[o0 + i] = ve;  eb[o0 + i] = f2bf(ve);
        of[o0 + i] = vo;  ob[o0 + i] = f2bf(vo);
    }
}

__global__ void k_tanhmul(const float* __restrict__ a, const float* __restrict__ x,
                          float* __restrict__ of, u16* __restrict__ ob, long n) {
    long i = (long)blockIdx.x * blockDim.x + threadIdx.x;
    long st = (long)gridDim.x * blockDim.x;
    for (; i < n; i += st) {
        float v = a[i] * tanhf(x[i]);
        of[i] = v; ob[i] = f2bf(v);
    }
}

__global__ void k_embskip(float* __restrict__ g, const float* __restrict__ emb,
                          const float* __restrict__ skip, long cnl) {
    long base = (long)blockIdx.y * cnl;
    long i = (long)blockIdx.x * blockDim.x + threadIdx.x;
    long st = (long)gridDim.x * blockDim.x;
    for (; i < cnl; i += st) g[base + i] = g[base + i] * emb[i] + skip[base + i];
}

__global__ void k_add(const float* __restrict__ a, const float* __restrict__ b,
                      float* __restrict__ o, long n) {
    long i = (long)blockIdx.x * blockDim.x + threadIdx.x;
    long st = (long)gridDim.x * blockDim.x;
    for (; i < n; i += st) o[i] = a[i] + b[i];
}

__global__ void k_interleave(const float* __restrict__ e, const float* __restrict__ o,
                             float* __restrict__ out, int Lh) {
    const int plane = NP * Lh;
    const long i0 = (long)blockIdx.y * plane;
    float* op = out + (long)blockIdx.y * (2 * plane);
    for (int i = blockIdx.x * 256 + threadIdx.x; i < plane; i += gridDim.x * 256) {
        int n = i / Lh, l = i - n * Lh;
        int ox = n * (2 * Lh) + 2 * l;
        op[ox]     = e[i0 + i];
        op[ox + 1] = o[i0 + i];
    }
}

__global__ void k_interleave_add(const float* __restrict__ e, const float* __restrict__ o,
                                 const float* __restrict__ x, float* __restrict__ out,
                                 int Lh) {
    const int plane = NREAL * Lh;
    const long pb = (long)blockIdx.y * (NP * Lh);
    const float* xp = x + (long)blockIdx.y * (NREAL * 2 * Lh);
    float* op = out + (long)blockIdx.y * (NREAL * 2 * Lh);
    for (int i = blockIdx.x * 256 + threadIdx.x; i < plane; i += gridDim.x * 256) {
        int n = i / Lh, l = i - n * Lh;
        int pi = n * Lh + l;
        int ox = n * (2 * Lh) + 2 * l;
        op[ox]     = e[pb + pi] + xp[ox];
        op[ox + 1] = o[pb + pi] + xp[ox + 1];
    }
}

// ---------------------------------------------------------------------------
// Host orchestration
// ---------------------------------------------------------------------------
namespace {

constexpr int Bn   = 32;
constexpr int Cc   = CC;
constexpr int Ltot = 48;

struct Arena {
    char*  base;
    size_t off;
    size_t cap;
    void* alloc(size_t bytes) {
        size_t a = (off + 255) & ~size_t(255);
        void* p = base + a;
        off = a + bytes;
        return p;
    }
};

struct LvlW {
    const u16*   tsw1[4]; const float* tsb1[4];
    const u16*   tsw2[4]; const float* tsb2[4];
    const u16*   g1w;     const float* g1b;
    const u16*   g2wA;    const u16*   g2wB;   const float* g2b;
    const u16*   adjf;    const u16*   adjd;
    const float* emb;    // padded [C][NP][Lh]
};

inline long long magic40(int d) { return ((1LL << 40) + d - 1) / d; }

inline dim3 ewgrid(long long n) {
    long long b = (n + 255) / 256;
    if (b > 16384) b = 16384;
    if (b < 1) b = 1;
    return dim3((unsigned)b);
}
inline dim3 plgrid(int plane, int planes) {
    int b = (plane + 255) / 256;
    return dim3((unsigned)b, (unsigned)planes);
}

void gemm(hipStream_t s,
          const u16* A, long sAb, int ldA,
          const u16* Bm, long sBb, int sBq, int sBk, int L2,
          float* Cf, u16* Cb, long sCb, int sCm, int sCq,
          const float* bias, int Mrows, int Ncol, int K, int act, int accum) {
    dim3 grid(Ncol / TN, Mrows / TM, Bn);
    k_wmma_gemm<<<grid, 256, 0, s>>>(A, sAb, ldA, Bm, sBb, sBq, sBk, L2, magic40(L2),
                                     Cf, Cb, sCb, sCm, sCq, bias, K, act, accum);
}

void conv_time(hipStream_t s, const u16* Wt, const float* bias, const u16* X,
               int Lin, int Lout, int KT, int toff, int clampF, int act,
               float* outF, u16* outB) {
    dim3 grid((NP * Lout) / TN, Cc / TM, Bn);
    k_wmma_conv<<<grid, 256, 0, s>>>(Wt, X, outF, outB, bias,
                                     Lin, Lout, magic40(Lout), KT, toff, clampF, act);
}

void dgcn(hipStream_t s, Arena a, const LvlW& w,
          const u16* tsB, const float* tsF, int Lh, float* outF) {
    const int  NL = NP * Lh;
    const long S  = (long)Bn * Cc * NL;
    u16* h  = (u16*)a.alloc((size_t)S * 2);
    u16* h1 = (u16*)a.alloc((size_t)S * 2);
    u16* h2 = (u16*)a.alloc((size_t)S * 2);

    gemm(s, w.g1w, 0, Cc,
         tsB, (long)Cc * NL, 0, NL, NL,
         nullptr, h, (long)Cc * NL, NL, 0,
         w.g1b, Cc, NL, Cc, 0, 0);

    gemm(s, w.adjf, 0, NP,
         h, (long)Cc * NL, NL, Lh, Lh,
         nullptr, h1, (long)Cc * NL, Lh, NL,
         nullptr, NP, Cc * Lh, NP, 0, 0);

    gemm(s, w.adjd, (long)NP * NP, NP,
         h1, (long)Cc * NL, NL, Lh, Lh,
         nullptr, h2, (long)Cc * NL, Lh, NL,
         nullptr, NP, Cc * Lh, NP, 0, 0);

    gemm(s, w.g2wA, 0, 2 * Cc,
         h1, (long)Cc * NL, 0, NL, NL,
         outF, nullptr, (long)Cc * NL, NL, 0,
         nullptr, Cc, NL, Cc, 0, 0);
    gemm(s, w.g2wB, 0, 2 * Cc,
         h2, (long)Cc * NL, 0, NL, NL,
         outF, nullptr, (long)Cc * NL, NL, 0,
         w.g2b, Cc, NL, Cc, 0, 1);

    dim3 g(ewgrid((long)Cc * NL).x, Bn);
    k_embskip<<<g, 256, 0, s>>>(outF, w.emb, tsF, (long)Cc * NL);
}

void tsconv(hipStream_t s, Arena a, const u16* xb, int Lh,
            const u16* w1, const float* b1, const u16* w2, const float* b2,
            float* outF, u16* outB) {
    const long Smid = (long)Bn * Cc * NP * (Lh + 2);
    u16* ct = (u16*)a.alloc((size_t)Smid * 2);
    conv_time(s, w1, b1, xb, Lh,     Lh + 2, 5, -3, 1, 1, nullptr, ct);
    conv_time(s, w2, b2, ct, Lh + 2, Lh,     3,  0, 0, 2, outF, outB);
}

void idgcn(hipStream_t s, Arena a, const float* x, int inN, int Lx, const LvlW& w,
           float* evenOut, float* oddOut) {
    const int  Lh = Lx / 2;
    const long S  = (long)Bn * Cc * NP * Lh;

    float* xe_f = (float*)a.alloc((size_t)S * 4);
    u16*   xe_b = (u16*)  a.alloc((size_t)S * 2);
    float* xo_f = (float*)a.alloc((size_t)S * 4);
    u16*   xo_b = (u16*)  a.alloc((size_t)S * 2);
    float* d_f  = (float*)a.alloc((size_t)S * 4);
    u16*   d_b  = (u16*)  a.alloc((size_t)S * 2);
    float* c_f  = (float*)a.alloc((size_t)S * 4);
    u16*   c_b  = (u16*)  a.alloc((size_t)S * 2);

    k_split<<<plgrid(NP * Lh, Bn * Cc), 256, 0, s>>>(x, inN, Lx, xe_f, xe_b, xo_f, xo_b);

    auto branch = [&](Arena t, const u16* inB, int bi, float* xout) {
        float* ts_f = (float*)t.alloc((size_t)S * 4);
        u16*   ts_b = (u16*)  t.alloc((size_t)S * 2);
        tsconv(s, t, inB, Lh, w.tsw1[bi], w.tsb1[bi], w.tsw2[bi], w.tsb2[bi], ts_f, ts_b);
        dgcn(s, t, w, ts_b, ts_f, Lh, xout);
    };

    { Arena t = a; float* x1 = (float*)t.alloc((size_t)S * 4);
      branch(t, xe_b, 0, x1);
      k_tanhmul<<<ewgrid(S), 256, 0, s>>>(xo_f, x1, d_f, d_b, S); }
    { Arena t = a; float* x2 = (float*)t.alloc((size_t)S * 4);
      branch(t, xo_b, 1, x2);
      k_tanhmul<<<ewgrid(S), 256, 0, s>>>(xe_f, x2, c_f, c_b, S); }
    { Arena t = a; float* x3 = (float*)t.alloc((size_t)S * 4);
      branch(t, c_b, 2, x3);
      k_add<<<ewgrid(S), 256, 0, s>>>(d_f, x3, oddOut, S); }
    { Arena t = a; float* x4 = (float*)t.alloc((size_t)S * 4);
      branch(t, d_b, 3, x4);
      k_add<<<ewgrid(S), 256, 0, s>>>(c_f, x4, evenOut, S); }
}

} // namespace

extern "C" void kernel_launch(void* const* d_in, const int* in_sizes, int n_in,
                              void* d_out, int out_size, void* d_ws, size_t ws_size,
                              hipStream_t stream) {
    const float* x     = (const float*)d_in[0];
    const float* adjf  = (const float*)d_in[1];
    const float* adjd  = (const float*)d_in[2];
    const float* tsw1  = (const float*)d_in[3];
    const float* tsb1  = (const float*)d_in[4];
    const float* tsw2  = (const float*)d_in[5];
    const float* tsb2  = (const float*)d_in[6];
    const float* g1w   = (const float*)d_in[7];
    const float* g1b   = (const float*)d_in[8];
    const float* g2w   = (const float*)d_in[9];
    const float* g2b   = (const float*)d_in[10];
    const float* embp[3] = { (const float*)d_in[11], (const float*)d_in[12],
                             (const float*)d_in[13] };
    const int embLh[3] = { 24, 12, 12 };

    Arena a{ (char*)d_ws, 0, ws_size };

    const long n_tsw1 = 3L * 4 * Cc * Cc * 5;
    const long n_tsw2 = 3L * 4 * Cc * Cc * 3;
    const long n_g1w  = 3L * Cc * Cc;
    const long n_g2w  = 3L * Cc * 2 * Cc;

    u16* tsw1_b = (u16*)a.alloc((size_t)n_tsw1 * 2);
    u16* tsw2_b = (u16*)a.alloc((size_t)n_tsw2 * 2);
    u16* g1w_b  = (u16*)a.alloc((size_t)n_g1w  * 2);
    u16* g2w_b  = (u16*)a.alloc((size_t)n_g2w  * 2);
    u16* adjf_t = (u16*)a.alloc((size_t)NP * NP * 2);
    u16* adjd_t = (u16*)a.alloc((size_t)Bn * NP * NP * 2);
    float* pemb[3];
    for (int l = 0; l < 3; ++l)
        pemb[l] = (float*)a.alloc((size_t)Cc * NP * embLh[l] * 4);

    k_conv_w<<<ewgrid(n_tsw1), 256, 0, stream>>>(tsw1, tsw1_b, 12, Cc, Cc, 5);
    k_conv_w<<<ewgrid(n_tsw2), 256, 0, stream>>>(tsw2, tsw2_b, 12, Cc, Cc, 3);
    k_f32_to_bf16<<<ewgrid(n_g1w), 256, 0, stream>>>(g1w, g1w_b, n_g1w);
    k_f32_to_bf16<<<ewgrid(n_g2w), 256, 0, stream>>>(g2w, g2w_b, n_g2w);
    k_adj_t<<<ewgrid((long)NP * NP), 256, 0, stream>>>(adjf, adjf_t, 1);
    k_adj_t<<<ewgrid((long)Bn * NP * NP), 256, 0, stream>>>(adjd, adjd_t, Bn);
    for (int l = 0; l < 3; ++l)
        k_pad_emb<<<plgrid(NP * embLh[l], Cc), 256, 0, stream>>>(embp[l], pemb[l], embLh[l]);

    LvlW lvl[3];
    for (int l = 0; l < 3; ++l) {
        for (int i = 0; i < 4; ++i) {
            lvl[l].tsw1[i] = tsw1_b + (long)(l * 4 + i) * Cc * Cc * 5;
            lvl[l].tsb1[i] = tsb1   + (long)(l * 4 + i) * Cc;
            lvl[l].tsw2[i] = tsw2_b + (long)(l * 4 + i) * Cc * Cc * 3;
            lvl[l].tsb2[i] = tsb2   + (long)(l * 4 + i) * Cc;
        }
        lvl[l].g1w  = g1w_b + (long)l * Cc * Cc;
        lvl[l].g1b  = g1b   + (long)l * Cc;
        lvl[l].g2wA = g2w_b + (long)l * Cc * 2 * Cc;
        lvl[l].g2wB = lvl[l].g2wA + Cc;
        lvl[l].g2b  = g2b   + (long)l * Cc;
        lvl[l].adjf = adjf_t;
        lvl[l].adjd = adjd_t;
        lvl[l].emb  = pemb[l];
    }

    const long S24 = (long)Bn * Cc * NP * 24;
    const long S12 = (long)Bn * Cc * NP * 12;

    float* e1 = (float*)a.alloc((size_t)S24 * 4);
    float* o1 = (float*)a.alloc((size_t)S24 * 4);
    float* e2 = (float*)a.alloc((size_t)S12 * 4);
    float* o2 = (float*)a.alloc((size_t)S12 * 4);
    float* e3 = (float*)a.alloc((size_t)S12 * 4);
    float* o3 = (float*)a.alloc((size_t)S12 * 4);
    float* cat1 = (float*)a.alloc((size_t)S24 * 4);
    float* cat2 = (float*)a.alloc((size_t)S24 * 4);

    idgcn(stream, a, x,  NREAL, Ltot, lvl[0], e1, o1);
    idgcn(stream, a, e1, NP,    24,   lvl[1], e2, o2);
    idgcn(stream, a, o1, NP,    24,   lvl[2], e3, o3);

    k_interleave<<<plgrid(NP * 12, Bn * Cc), 256, 0, stream>>>(e2, o2, cat1, 12);
    k_interleave<<<plgrid(NP * 12, Bn * Cc), 256, 0, stream>>>(e3, o3, cat2, 12);
    k_interleave_add<<<plgrid(NREAL * 24, Bn * Cc), 256, 0, stream>>>(
        cat1, cat2, x, (float*)d_out, 24);
}
